// ProteinFeaturesNA_34351148434195
// MI455X (gfx1250) — matrix operand order repaired and checked by reading, not compiled, not run on hardware
//
#include <hip/hip_runtime.h>
#include <hip/hip_bf16.h>
#include <stdint.h>

typedef __attribute__((ext_vector_type(16))) __bf16 bf16x16;
typedef __attribute__((ext_vector_type(8)))  float  v8f;
typedef __attribute__((ext_vector_type(16))) unsigned short ushort16;
typedef __attribute__((ext_vector_type(8)))  unsigned short ushort8;

namespace {
constexpr int LRES   = 512;
constexpr int NATM   = 24;
constexpr int AUGA   = 26;          // 24 atoms + Cb + N_na
constexpr int TOPK   = 30;
constexpr int NPAIR  = LRES * TOPK; // 15360
constexpr int EDGE_IN = 10832;      // 16 positional + 26*26*16 RBF
constexpr int NCELL  = AUGA * AUGA; // 676
constexpr int KPAD   = 10880;       // 170 * 64, zero padded
constexpr int KSTEPS = KPAD / 64;   // 170
constexpr int CH     = 128;
constexpr int V_SIZE = LRES * CH;       // 65536 floats
constexpr int E_SIZE = NPAIR * CH;      // 1966080 floats
// workspace layout (floats / ints / ushorts)
constexpr int WS_AUGX  = 0;                       // 512*26*3
constexpr int WS_AUGXM = WS_AUGX + LRES*AUGA*3;   // 512*26
constexpr int WS_XC    = WS_AUGXM + LRES*AUGA;    // 512*3
constexpr int WS_EIDX  = WS_XC + LRES*3;          // 15360 ints
constexpr int WS_WPACK = WS_EIDX + NPAIR;         // then ushorts (byte off 280576, 32B aligned)
}

__device__ __forceinline__ unsigned short f2bf(float f) {
  unsigned int u = __float_as_uint(f);
  u += 0x7FFFu + ((u >> 16) & 1u);   // round-to-nearest-even
  return (unsigned short)(u >> 16);
}

// ---------------------------------------------------------------- prep
__global__ void pf_prep_kernel(const float* __restrict__ X,
                               const float* __restrict__ Xm,
                               const float* __restrict__ pmask,
                               const float* __restrict__ dmask,
                               const float* __restrict__ rmask,
                               float* __restrict__ augX,
                               float* __restrict__ augXm,
                               float* __restrict__ Xc) {
  int i = blockIdx.x * blockDim.x + threadIdx.x;
  if (i >= LRES) return;
  const float* xr = X + (size_t)i * NATM * 3;
  float* ax = augX + (size_t)i * AUGA * 3;
  #pragma unroll 4
  for (int a = 0; a < NATM; ++a) {
    ax[a*3+0] = xr[a*3+0]; ax[a*3+1] = xr[a*3+1]; ax[a*3+2] = xr[a*3+2];
  }
  // Cb from N(0), CA(1), C(2)
  {
    float nx=xr[0],ny=xr[1],nz=xr[2], cax=xr[3],cay=xr[4],caz=xr[5], cx=xr[6],cy=xr[7],cz=xr[8];
    float bx=cax-nx, by=cay-ny, bz=caz-nz;
    float ex=cx-cax, ey=cy-cay, ez=cz-caz;
    float axx=by*ez-bz*ey, ayy=bz*ex-bx*ez, azz=bx*ey-by*ex;
    const float wa=-0.58273431f, wb=0.56802827f, wc=-0.54067466f;
    ax[24*3+0]=wa*axx+wb*bx+wc*ex+cax;
    ax[24*3+1]=wa*ayy+wb*by+wc*ey+cay;
    ax[24*3+2]=wa*azz+wb*bz+wc*ez+caz;
  }
  // N_na from O4'(11), C1'(15), C2'(14)
  {
    float nx=xr[11*3],ny=xr[11*3+1],nz=xr[11*3+2];
    float cax=xr[15*3],cay=xr[15*3+1],caz=xr[15*3+2];
    float cx=xr[14*3],cy=xr[14*3+1],cz=xr[14*3+2];
    float bx=cax-nx, by=cay-ny, bz=caz-nz;
    float ex=cx-cax, ey=cy-cay, ez=cz-caz;
    float axx=by*ez-bz*ey, ayy=bz*ex-bx*ez, azz=bx*ey-by*ex;
    const float wa=-0.56967352f, wb=0.51055973f, wc=-0.53122153f;
    ax[25*3+0]=wa*axx+wb*bx+wc*ex+cax;
    ax[25*3+1]=wa*ayy+wb*by+wc*ey+cay;
    ax[25*3+2]=wa*azz+wb*bz+wc*ez+caz;
  }
  float* am = augXm + (size_t)i * AUGA;
  #pragma unroll 4
  for (int a = 0; a < NATM; ++a) am[a] = Xm[(size_t)i*NATM + a];
  am[24] = pmask[i];
  am[25] = dmask[i] + rmask[i];
  // Xc = CA + na_ref(15)
  Xc[i*3+0] = xr[3]  + xr[15*3+0];
  Xc[i*3+1] = xr[4]  + xr[15*3+1];
  Xc[i*3+2] = xr[5]  + xr[15*3+2];
}

// ------------------------------------------------- W_edge -> bf16 fragment pack
// block(kc, nt): [32 lanes][16 vals]; lane<16: n=nt*16+lane, k=kc*32+idx
//                lane>=16: n=nt*16+(lane-16), k=kc*32+16+idx   (contiguous K per lane half, same order as the A reads)
__global__ void pf_packw_kernel(const float* __restrict__ We,
                                unsigned short* __restrict__ wpack) {
  int g = blockIdx.x * blockDim.x + threadIdx.x;
  int idx  =  g        & 15;
  int lane = (g >> 4)  & 31;
  int nt   = (g >> 9)  & 7;
  int kc   =  g >> 12;
  int n = nt * 16 + (lane & 15);
  int k = kc * 32 + ((lane >> 4) << 4) + idx;
  float v = (k < EDGE_IN) ? We[(size_t)n * EDGE_IN + k] : 0.0f;
  wpack[g] = f2bf(v);
}

// ---------------------------------------------------------------- top-k
__global__ void pf_topk_kernel(const float* __restrict__ Xc,
                               const float* __restrict__ mask,
                               int* __restrict__ eidx,
                               float* __restrict__ outEidx) {
  __shared__ float dadj[LRES];
  __shared__ float redf[256];
  __shared__ unsigned long long redk[256];
  const int i = blockIdx.x, tid = threadIdx.x;
  const float xi = Xc[i*3], yi = Xc[i*3+1], zi = Xc[i*3+2];
  const float mi = mask[i];
  float lmax = 0.0f;
  for (int j = tid; j < LRES; j += 256) {
    float dx = xi - Xc[j*3], dy = yi - Xc[j*3+1], dz = zi - Xc[j*3+2];
    float D = mi * mask[j] * sqrtf(dx*dx + dy*dy + dz*dz + 1e-6f);
    dadj[j] = D;
    lmax = fmaxf(lmax, D);
  }
  redf[tid] = lmax; __syncthreads();
  for (int s = 128; s > 0; s >>= 1) {
    if (tid < s) redf[tid] = fmaxf(redf[tid], redf[tid + s]);
    __syncthreads();
  }
  const float Dmax = redf[0];
  __syncthreads();
  for (int j = tid; j < LRES; j += 256) {
    float m2 = mi * mask[j];
    dadj[j] = dadj[j] + (1.0f - m2) * Dmax;
  }
  __syncthreads();
  for (int r = 0; r < TOPK; ++r) {
    unsigned long long best = ~0ull;
    for (int j = tid; j < LRES; j += 256) {
      unsigned long long key =
          (((unsigned long long)__float_as_uint(dadj[j])) << 32) | (unsigned)j;
      best = key < best ? key : best;
    }
    redk[tid] = best; __syncthreads();
    for (int s = 128; s > 0; s >>= 1) {
      if (tid < s && redk[tid + s] < redk[tid]) redk[tid] = redk[tid + s];
      __syncthreads();
    }
    if (tid == 0) {
      int jw = (int)(redk[0] & 0xffffffffu);
      eidx[i*TOPK + r] = jw;
      outEidx[i*TOPK + r] = (float)jw;
      dadj[jw] = __uint_as_float(0x7F800000u);   // +inf: remove from candidates
    }
    __syncthreads();
  }
}

// ---------------------------------------------------------------- node (V)
__global__ void pf_node_kernel(const int* __restrict__ ptype,
                               const float* __restrict__ Wn,
                               const float* __restrict__ gn,
                               const float* __restrict__ bn,
                               float* __restrict__ outV) {
  __shared__ float rs[CH], rq[CH];
  const int i = blockIdx.x, c = threadIdx.x;
  const int t = ptype[i];
  float v = Wn[c*3 + t];
  rs[c] = v; rq[c] = v*v; __syncthreads();
  for (int s = 64; s > 0; s >>= 1) {
    if (c < s) { rs[c] += rs[c+s]; rq[c] += rq[c+s]; }
    __syncthreads();
  }
  float mean = rs[0] * (1.0f/CH);
  float var  = rq[0] * (1.0f/CH) - mean*mean;
  float inv  = rsqrtf(var + 1e-5f);
  outV[(size_t)i*CH + c] = (v - mean) * inv * gn[c] + bn[c];
}

// ---------------------------------------------------------------- edge GEMM
// 16 pairs x 128 channels per block, 8 waves, fused feature gen + WMMA + LN.
__global__ __launch_bounds__(256)
void pf_edge_kernel(const float* __restrict__ augX,
                    const float* __restrict__ augXm,
                    const int*   __restrict__ eidx,
                    const int*   __restrict__ Ridx,
                    const int*   __restrict__ chain,
                    const float* __restrict__ Wpos,
                    const float* __restrict__ bpos,
                    const unsigned short* __restrict__ wpack,
                    const float* __restrict__ ge,
                    const float* __restrict__ be,
                    float* __restrict__ outE) {
  __shared__ __align__(16) float4 sPI[16][AUGA];          // (x,y,z,mask) i-side
  __shared__ __align__(16) float4 sPJ[16][AUGA];          // (x,y,z,mask) j-side
  __shared__ int   sD[16];
  __shared__ __align__(16) unsigned short sA[2][16][72];  // 64 bf16 + pad, 144B stride
  __shared__ float sE[16][132];                           // padded C spill

  const int tid  = threadIdx.x;
  const int wave = tid >> 5;
  const int lane = tid & 31;
  const int p0   = blockIdx.x * 16;

  // ---- stage pair atoms + masks as packed float4
  for (int idx = tid; idx < 16 * AUGA; idx += 256) {
    int m = idx / AUGA, a = idx % AUGA;
    int p = p0 + m;
    int i = p / TOPK;
    int j = eidx[p];
    const float* xi = augX + ((size_t)i*AUGA + a) * 3;
    const float* xj = augX + ((size_t)j*AUGA + a) * 3;
    sPI[m][a] = make_float4(xi[0], xi[1], xi[2], augXm[(size_t)i*AUGA + a]);
    sPJ[m][a] = make_float4(xj[0], xj[1], xj[2], augXm[(size_t)j*AUGA + a]);
  }
  if (tid < 16) {
    int p = p0 + tid;
    int i = p / TOPK;
    int j = eidx[p];
    int off = Ridx[i] - Ridx[j];
    int ch  = (chain[i] == chain[j]);
    int dd  = off + 32; dd = dd < 0 ? 0 : (dd > 64 ? 64 : dd);
    sD[tid] = ch ? dd : 65;
  }
  __syncthreads();

  const int mrow = tid >> 4;       // pair row 0..15 (feature gen)
  const int r    = tid & 15;       // rbf / channel index 0..15
  const float mu = 2.0f + (float)r * (20.0f / 15.0f);

  // positional channel value (only feature with cellIdx == -1)
  const float posval = Wpos[r * 66 + sD[mrow]] + bpos[r];

  // incremental cell state: thread visits cells strictly sequentially
  // (cell = 4*s + c - 1); i-side atom cached in registers, reloaded on wrap.
  int cellIdx = -1, a1 = 0, a2 = 0;
  float4 pi = sPI[mrow][0];

  v8f acc = {};
  for (int s = 0; s < KSTEPS; ++s) {
    const int buf = s & 1;
    unsigned short* arow = &sA[buf][mrow][0];
    #pragma unroll
    for (int c = 0; c < 4; ++c) {
      float val;
      if (cellIdx < 0) {
        val = posval;
      } else if (cellIdx < NCELL) {
        float4 pj = sPJ[mrow][a2];           // one broadcast ds_load_b128
        float dx = pi.x - pj.x;
        float dy = pi.y - pj.y;
        float dz = pi.z - pj.z;
        float d  = sqrtf(fmaf(dx, dx, fmaf(dy, dy, fmaf(dz, dz, 1e-6f))));
        float t  = (d - mu) * 0.8f;          // / sigma(=1.25)
        val = pi.w * pj.w * __expf(-t * t);
      } else {
        val = 0.0f;                          // K padding
      }
      arow[c * 16 + r] = f2bf(val);
      // advance sequential (a1,a2) state
      if (cellIdx >= 0) {
        if (++a2 == AUGA) {
          a2 = 0;
          if (++a1 < AUGA) pi = sPI[mrow][a1];
        }
      }
      ++cellIdx;
    }
    __syncthreads();

    const int khoff = (lane >> 4) * 16;  // contiguous K per lane half (matches wpack)
    #pragma unroll
    for (int q = 0; q < 2; ++q) {
      const unsigned short* abase = &sA[buf][lane & 15][q * 32];
      union { ushort8 h[2]; ushort16 w; } ua;
      ua.h[0] = *(const ushort8*)(abase + khoff);        // K 0..7 / 16..23
      ua.h[1] = *(const ushort8*)(abase + 8 + khoff);    // K 8..15 / 24..31
      bf16x16 af = __builtin_bit_cast(bf16x16, ua.w);

      size_t boff = (((size_t)(s * 2 + q) * 8 + wave) * 32 + lane) * 16;
      ushort16 bw = *(const ushort16*)(wpack + boff);
      bf16x16 bf = __builtin_bit_cast(bf16x16, bw);

      acc = __builtin_amdgcn_wmma_f32_16x16x32_bf16(
          false, af, false, bf, (short)0, acc, false, false);
    }
    // single barrier per step; double buffer covers the gen->read hazard
  }
  __syncthreads();

  // ---- spill C (16x16 per wave) to LDS: row = v + 8*(lane>>4), col = wave*16 + lane&15
  {
    const int col = wave * 16 + (lane & 15);
    const int rh  = (lane >> 4) * 8;
    #pragma unroll
    for (int v = 0; v < 8; ++v) sE[v + rh][col] = acc[v];
  }
  __syncthreads();

  // ---- fused LayerNorm over 128 channels, 16 lanes per row
  {
    const int row = tid >> 4, seg = tid & 15;
    float vals[8], s1 = 0.0f, s2 = 0.0f;
    #pragma unroll
    for (int u = 0; u < 8; ++u) {
      float x = sE[row][seg * 8 + u];
      vals[u] = x; s1 += x; s2 += x * x;
    }
    #pragma unroll
    for (int o = 8; o >= 1; o >>= 1) {
      s1 += __shfl_xor(s1, o, 16);
      s2 += __shfl_xor(s2, o, 16);
    }
    float mean = s1 * (1.0f / CH);
    float var  = s2 * (1.0f / CH) - mean * mean;
    float inv  = rsqrtf(var + 1e-5f);
    const int p = p0 + row;
    __align__(16) float outv[8];
    #pragma unroll
    for (int u = 0; u < 8; ++u) {
      int c = seg * 8 + u;
      outv[u] = (vals[u] - mean) * inv * ge[c] + be[c];
    }
    float4* dst = (float4*)(outE + (size_t)p * CH + seg * 8);
    dst[0] = *(const float4*)&outv[0];
    dst[1] = *(const float4*)&outv[4];
  }
}

// ---------------------------------------------------------------- launch
extern "C" void kernel_launch(void* const* d_in, const int* in_sizes, int n_in,
                              void* d_out, int out_size, void* d_ws, size_t ws_size,
                              hipStream_t stream) {
  const float* X      = (const float*)d_in[0];
  const float* mask   = (const float*)d_in[1];
  const int*   Ridx   = (const int*)  d_in[2];
  const int*   chain  = (const int*)  d_in[3];
  const float* Xm     = (const float*)d_in[4];
  const float* pmask  = (const float*)d_in[5];
  const float* dmask  = (const float*)d_in[6];
  const float* rmask  = (const float*)d_in[7];
  const int*   ptype  = (const int*)  d_in[8];
  const float* Wpos   = (const float*)d_in[9];
  const float* bpos   = (const float*)d_in[10];
  const float* Wedge  = (const float*)d_in[11];
  const float* Wnode  = (const float*)d_in[12];
  const float* ge     = (const float*)d_in[13];
  const float* be     = (const float*)d_in[14];
  const float* gn     = (const float*)d_in[15];
  const float* bn     = (const float*)d_in[16];

  float* ws        = (float*)d_ws;
  float* augX      = ws + WS_AUGX;
  float* augXm     = ws + WS_AUGXM;
  float* Xc        = ws + WS_XC;
  int*   eidx      = (int*)(ws + WS_EIDX);
  unsigned short* wpack = (unsigned short*)(ws + WS_WPACK); // 32B aligned

  float* outV    = (float*)d_out;
  float* outE    = outV + V_SIZE;
  float* outEidx = outV + V_SIZE + E_SIZE;

  pf_prep_kernel<<<(LRES + 63) / 64, 64, 0, stream>>>(
      X, Xm, pmask, dmask, rmask, augX, augXm, Xc);

  pf_packw_kernel<<<(KPAD / 32) * 8 * 32 * 16 / 256, 256, 0, stream>>>(Wedge, wpack);

  pf_topk_kernel<<<LRES, 256, 0, stream>>>(Xc, mask, eidx, outEidx);

  pf_node_kernel<<<LRES, CH, 0, stream>>>(ptype, Wnode, gn, bn, outV);

  pf_edge_kernel<<<NPAIR / 16, 256, 0, stream>>>(
      augX, augXm, eidx, Ridx, chain, Wpos, bpos, wpack, ge, be, outE);
}
